// ChessNetwork_19061064860377
// MI455X (gfx1250) — compile-verified, matrix operand order checked by reading, not run on hardware
//
#include <hip/hip_runtime.h>
#include <math.h>

typedef __attribute__((ext_vector_type(16))) _Float16 v16h;
typedef __attribute__((ext_vector_type(8)))  _Float16 v8h;
typedef __attribute__((ext_vector_type(8)))  float    v8f;
typedef __attribute__((ext_vector_type(4)))  unsigned int u32x4;
typedef __attribute__((ext_vector_type(8)))  int      i32x8;
typedef __attribute__((ext_vector_type(4)))  int      i32x4;

#define NBATCH 16384
#define EPSBN  1e-5f

// packed f16 weight region layout (halfs):
//   conv0:      [0, 5120)          5 steps * 1024
//   convs i:    [5120 + i*9216)    9 steps * 1024, i = 0..5
//   pfc:        [60416, 156672)    752 rows * 128
#define WPK_CONV0 0
#define WPK_CONVS 5120
#define WPK_PFC   60416
#define WPK_TOTAL 156672

__device__ __forceinline__ v16h cat16(v8h lo, v8h hi) {
    return __builtin_shufflevector(lo, hi, 0,1,2,3,4,5,6,7,8,9,10,11,12,13,14,15);
}
__device__ __forceinline__ v16h load_frag(const _Float16* p) {
    v8h lo = *(const v8h*)(p);
    v8h hi = *(const v8h*)(p + 16);
    return cat16(lo, hi);
}

// ---------------------------------------------------------------------------
// Tensor Data Mover: 1-D tile of `nelem` f16 elements, global -> LDS.
// D# group0: count=1, lds_addr, global_addr, type=2.  group1: data_size=2B,
// tensor_dim0=nelem, tensor_dim1=1, tile_dim0=nelem.
// ---------------------------------------------------------------------------
__device__ __forceinline__ void tdm_load_lds(_Float16* ldst, const _Float16* gsrc,
                                             int nelem) {
    unsigned long long ga = (unsigned long long)(uintptr_t)gsrc;
    unsigned int lo = (unsigned int)(uintptr_t)ldst;   // LDS byte offset (addr[31:0])
    u32x4 g0 = { 1u,                                   // [1:0] count = 1
                 lo,                                    // [63:32] lds_addr
                 (unsigned int)ga,                      // [95:64] global_addr lo
                 (unsigned int)((ga >> 32) & 0x1FFFFFFu) | (2u << 30) }; // hi + type=2
    unsigned int ne = (unsigned int)nelem;
    i32x8 g1 = { (int)(1u << 16),                      // data_size = 1 (2 bytes)
                 (int)((ne & 0xFFFFu) << 16),          // tensor_dim0[15:0]  @ bits 63:48
                 (int)((ne >> 16) | (1u << 16)),       // tensor_dim0 hi, tensor_dim1 = 1
                 (int)(ne << 16),                      // tile_dim0          @ bits 127:112
                 0,                                    // tile_dim1/2 = 0 (unused)
                 (int)ne,                              // tensor_dim0_stride
                 0, 0 };
    i32x4 z4 = {};
#if __clang_major__ >= 23
    i32x8 z8 = {};
    __builtin_amdgcn_tensor_load_to_lds(g0, g1, z4, z4, z8, 0);
#else
    __builtin_amdgcn_tensor_load_to_lds(g0, g1, z4, z4, 0);
#endif
}

// ---------------------------------------------------------------------------
// init: zero stats slots, identity gabe slot 0
// ---------------------------------------------------------------------------
__global__ void init_ws(float* __restrict__ stats, float* __restrict__ gabe0) {
    int t = threadIdx.x;
    for (int i = t; i < 9 * 64; i += 256) stats[i] = 0.0f;
    if (t < 32) { gabe0[t] = 1.0f; gabe0[32 + t] = 0.0f; }
}

// ---------------------------------------------------------------------------
// pack all GEMM weights into K-packed f16 images (consumed via TDM / b128)
// ---------------------------------------------------------------------------
__global__ __launch_bounds__(256) void pack_weights(
    const float* __restrict__ c0w,   // (32, 9, 3, 3)
    const float* __restrict__ csw,   // (6, 32, 32, 3, 3)
    const float* __restrict__ pfw,   // (750, 128)
    _Float16* __restrict__ wp)
{
    int idx = blockIdx.x * 256 + threadIdx.x;
    if (idx < 5120) {                       // conv0: [s][n][k], taps packed 2/step
        int s = idx >> 10, r = idx & 1023;
        int n = r >> 5, k = r & 31;
        int tap = 2 * s + (k >> 4), c = k & 15;
        float v = (tap < 9 && c < 9) ? c0w[(n * 9 + c) * 9 + tap] : 0.0f;
        wp[idx] = (_Float16)v;
    } else if (idx < WPK_PFC) {             // convs: [layer][s][n][k], k = in-ch
        int t = idx - 5120;
        int layer = t / 9216, r = t - layer * 9216;
        int s = r >> 10, rr = r & 1023;
        int n = rr >> 5, k = rr & 31;
        wp[idx] = (_Float16)csw[((size_t)layer * 1024 + n * 32 + k) * 9 + s];
    } else if (idx < WPK_TOTAL) {           // pfc: [n(752)][k(128)], zero padded
        int t = idx - WPK_PFC;
        int n = t >> 7, k = t & 127;
        wp[idx] = (_Float16)((n < 750) ? pfw[n * 128 + k] : 0.0f);
    }
}

// ---------------------------------------------------------------------------
// bn finalize: stats(sum, sumsq) -> (ga, be) s.t. BN(y) = y*ga + be
// ---------------------------------------------------------------------------
__global__ void bn_finalize(const float* __restrict__ stats,
                            const float* __restrict__ scale,
                            const float* __restrict__ bias,
                            int nch, float count, float* __restrict__ gabe) {
    int c = threadIdx.x;
    if (c < 32) {
        float ga = 0.0f, be = 0.0f;
        if (c < nch) {
            float m   = stats[c] / count;
            float var = stats[32 + c] / count - m * m;
            ga = rsqrtf(var + EPSBN) * scale[c];
            be = bias[c] - m * ga;
        }
        gabe[c] = ga; gabe[32 + c] = be;
    }
}

// ---------------------------------------------------------------------------
// conv 3x3 SAME as implicit GEMM via v_wmma_f32_16x16x32_f16.
// MODE 1: 32 in-ch, 9 K-steps (one tap, K = channel).  MODE 0: conv0, 9 in-ch
// padded to 16, two taps per K-step (5 steps).  BN+ReLU of the *previous*
// layer applied on load.  Weights arrive pre-packed f16 via a single TDM op.
// block = 256 threads = 8 waves = 2 images (4 M-tiles of 16 positions each).
// ---------------------------------------------------------------------------
template <int MODE>
__global__ __launch_bounds__(256) void conv_wmma(
    const float* __restrict__ inF, const _Float16* __restrict__ inH,
    const float* __restrict__ gabe,
    const _Float16* __restrict__ wpk,
    _Float16* __restrict__ yout, float* __restrict__ stats)
{
    constexpr int NSTEPS = MODE ? 9 : 5;
    constexpr int INCH   = MODE ? 32 : 10;
    constexpr int CIN    = MODE ? 32 : 9;

    __shared__ __align__(16) _Float16 wlds[9 * 1024];   // [step][outn(32)][k(32)]
    __shared__ __align__(16) _Float16 alds[2 * 3200];   // [img][cell(10x10)][c(32)]
    __shared__ float lsum[32], lsq[32];

    const int tid = threadIdx.x;
    const int imgBase = blockIdx.x * 2;

    // ---- weights: one async TDM transfer into LDS ----
    if (tid == 0) tdm_load_lds(wlds, wpk, NSTEPS * 1024);

    // ---- zero the 36 halo cells per image (vector LDS stores) ----
    for (int idx = tid; idx < 2 * 144; idx += 256) {      // 36 cells * 4 v8h
        int img = idx / 144, r = idx - img * 144;
        int ci = r >> 2, part = r & 3;
        int yy, xx;
        if (ci < 10)      { yy = 0;       xx = ci; }
        else if (ci < 20) { yy = 9;       xx = ci - 10; }
        else if (ci < 28) { yy = ci - 19; xx = 0; }
        else              { yy = ci - 27; xx = 9; }
        v8h z = {};
        ((v8h*)(alds + img * 3200 + (yy * 10 + xx) * 32))[part] = z;
    }

    // ---- interior fill, pos-fastest for coalesced global loads ----
    for (int idx = tid; idx < 4096; idx += 256) {
        int pos = idx & 63, c = (idx >> 6) & 31, img = idx >> 11;
        float a = 0.0f;
        if (c < CIN) {
            size_t gi = ((size_t)(imgBase + img) * INCH + c) * 64 + pos;
            float raw = MODE ? (float)inH[gi] : inF[gi];
            a = raw * gabe[c] + gabe[32 + c];
            if (MODE) a = fmaxf(a, 0.0f);   // ReLU for all but the raw board input
        }
        alds[img * 3200 + ((pos >> 3) + 1) * 320 + ((pos & 7) + 1) * 32 + c] = (_Float16)a;
    }
    if (tid < 32) { lsum[tid] = 0.0f; lsq[tid] = 0.0f; }
    __builtin_amdgcn_s_wait_tensorcnt(0);
    __syncthreads();

    const int wave  = tid >> 5;
    const int lane  = tid & 31;
    const int img   = wave >> 2;       // 0..1
    const int mtile = wave & 3;        // 0..3
    const int m     = lane & 15;
    const int khigh = lane >> 4;
    const int pos   = mtile * 16 + m;
    const int py = pos >> 3, px = pos & 7;

    v8f acc0 = {}; v8f acc1 = {};
    const _Float16* abase = alds + img * 3200;

    #pragma unroll
    for (int s = 0; s < NSTEPS; ++s) {
        v16h afrag;
        if (MODE) {
            int dy = s / 3 - 1, dx = s % 3 - 1;
            afrag = load_frag(abase + ((py + dy + 1) * 10 + (px + dx + 1)) * 32 + khigh * 8);
        } else {
            int t0 = 2 * s, t1 = 2 * s + 1;
            int dy0 = t0 / 3 - 1, dx0 = t0 % 3 - 1;
            v8h alo = *(const v8h*)(abase + ((py + dy0 + 1) * 10 + (px + dx0 + 1)) * 32 + khigh * 8);
            v8h ahi = {};
            if (t1 < 9) {
                int dy1 = t1 / 3 - 1, dx1 = t1 % 3 - 1;
                ahi = *(const v8h*)(abase + ((py + dy1 + 1) * 10 + (px + dx1 + 1)) * 32 + khigh * 8);
            }
            afrag = cat16(alo, ahi);
        }
        const _Float16* wb = wlds + s * 1024;
        v16h b0 = load_frag(wb + m * 32 + khigh * 8);
        acc0 = __builtin_amdgcn_wmma_f32_16x16x32_f16(false, afrag, false, b0,
                                                      (short)0, acc0, false, false);
        v16h b1 = load_frag(wb + (16 + m) * 32 + khigh * 8);
        acc1 = __builtin_amdgcn_wmma_f32_16x16x32_f16(false, afrag, false, b1,
                                                      (short)0, acc1, false, false);
    }

    // ---- write raw f16 y + batch stats (D layout: lane = N, vgpr i -> M) ----
    const size_t imgG = imgBase + img;
    float s0 = 0, q0 = 0, s1 = 0, q1 = 0;
    #pragma unroll
    for (int i = 0; i < 8; ++i) {
        int opos = mtile * 16 + i + khigh * 8;
        float v0 = acc0[i], v1 = acc1[i];
        yout[(imgG * 32 + m) * 64 + opos]        = (_Float16)v0;
        yout[(imgG * 32 + (16 + m)) * 64 + opos] = (_Float16)v1;
        s0 += v0; q0 += v0 * v0; s1 += v1; q1 += v1 * v1;
    }
    atomicAdd(&lsum[m], s0);        atomicAdd(&lsq[m], q0);
    atomicAdd(&lsum[16 + m], s1);   atomicAdd(&lsq[16 + m], q1);
    __syncthreads();
    if (tid < 32) {
        atomicAdd(&stats[tid], lsum[tid]);
        atomicAdd(&stats[32 + tid], lsq[tid]);
    }
}

// ---------------------------------------------------------------------------
// 1x1 head convs (policy 32->2, value 32->1), fused, with batch stats
// ---------------------------------------------------------------------------
__global__ __launch_bounds__(256) void head_conv(
    const _Float16* __restrict__ h, const float* __restrict__ gabe7,
    const float* __restrict__ polw, const float* __restrict__ valw,
    float* __restrict__ p_raw, float* __restrict__ v_raw,
    float* __restrict__ statsP, float* __restrict__ statsV)
{
    __shared__ float red[6];
    int tid = threadIdx.x;
    if (tid < 6) red[tid] = 0.0f;
    __syncthreads();
    size_t gidx = (size_t)blockIdx.x * 256 + tid;   // img*64 + pos
    size_t img  = gidx >> 6;
    int    pos  = (int)(gidx & 63);
    float p0 = 0, p1 = 0, vv = 0;
    #pragma unroll
    for (int c = 0; c < 32; ++c) {
        float raw = (float)h[(img * 32 + c) * 64 + pos];
        float a = fmaxf(raw * gabe7[c] + gabe7[32 + c], 0.0f);
        p0 += a * polw[c]; p1 += a * polw[32 + c]; vv += a * valw[c];
    }
    p_raw[img * 128 + pos]      = p0;
    p_raw[img * 128 + 64 + pos] = p1;
    v_raw[img * 64 + pos]       = vv;
    atomicAdd(&red[0], p0); atomicAdd(&red[1], p0 * p0);
    atomicAdd(&red[2], p1); atomicAdd(&red[3], p1 * p1);
    atomicAdd(&red[4], vv); atomicAdd(&red[5], vv * vv);
    __syncthreads();
    if (tid == 0) {
        atomicAdd(&statsP[0], red[0]); atomicAdd(&statsP[32], red[1]);
        atomicAdd(&statsP[1], red[2]); atomicAdd(&statsP[33], red[3]);
    }
    if (tid == 1) {
        atomicAdd(&statsV[0], red[4]); atomicAdd(&statsV[32], red[5]);
    }
}

// ---------------------------------------------------------------------------
// policy logits: (16384 x 128) @ (128 x 752pad) + bias via WMMA.
// block = one M-tile (16 images); 8 waves split the 47 N-tiles.
// B fragments come straight from packed f16 weights (two b128 loads each).
// ---------------------------------------------------------------------------
__global__ __launch_bounds__(256) void policy_wmma(
    const float* __restrict__ p_raw, const float* __restrict__ gabe8,
    const _Float16* __restrict__ pfcH, const float* __restrict__ pfcb,
    float* __restrict__ out)
{
    __shared__ __align__(16) _Float16 apanel[16 * 128];
    int tid = threadIdx.x;
    int mt  = blockIdx.x;
    for (int idx = tid; idx < 2048; idx += 256) {
        int mrow = idx >> 7;
        int f    = idx & 127;
        int c    = f >> 6;
        float raw = p_raw[(size_t)(mt * 16 + mrow) * 128 + f];
        apanel[idx] = (_Float16)fmaxf(raw * gabe8[c] + gabe8[32 + c], 0.0f);
    }
    __syncthreads();
    int wave = tid >> 5, lane = tid & 31;
    int m = lane & 15, khigh = lane >> 4;
    for (int nt = wave; nt < 47; nt += 8) {
        v8f acc = {};
        int n = nt * 16 + m;
        #pragma unroll
        for (int s = 0; s < 4; ++s) {
            v16h afrag = load_frag(apanel + m * 128 + s * 32 + khigh * 8);
            v16h bfrag = load_frag(pfcH + n * 128 + s * 32 + khigh * 8);
            acc = __builtin_amdgcn_wmma_f32_16x16x32_f16(false, afrag, false, bfrag,
                                                         (short)0, acc, false, false);
        }
        if (n < 750) {
            float bias = pfcb[n];
            #pragma unroll
            for (int i = 0; i < 8; ++i) {
                int img = mt * 16 + i + khigh * 8;
                out[(size_t)img * 750 + n] = acc[i] + bias;
            }
        }
    }
}

// ---------------------------------------------------------------------------
// value MLP: 64 -> 128 (relu) -> 1 (tanh). One block per image.
// ---------------------------------------------------------------------------
__global__ __launch_bounds__(128) void value_fc(
    const float* __restrict__ v_raw, const float* __restrict__ gabe9,
    const float* __restrict__ w1, const float* __restrict__ b1,
    const float* __restrict__ w2, const float* __restrict__ b2,
    float* __restrict__ out)
{
    __shared__ float a[64];
    __shared__ float red[128];
    int tid = threadIdx.x;
    size_t img = blockIdx.x;
    if (tid < 64) {
        float raw = v_raw[img * 64 + tid];
        a[tid] = fmaxf(raw * gabe9[0] + gabe9[32], 0.0f);
    }
    __syncthreads();
    float hsum = b1[tid];
    #pragma unroll 8
    for (int k = 0; k < 64; ++k) hsum += a[k] * w1[tid * 64 + k];
    hsum = fmaxf(hsum, 0.0f);
    red[tid] = hsum * w2[tid];
    __syncthreads();
    for (int s = 64; s > 0; s >>= 1) {
        if (tid < s) red[tid] += red[tid + s];
        __syncthreads();
    }
    if (tid == 0) out[img] = tanhf(red[0] + b2[0]);
}

// ---------------------------------------------------------------------------
// piece_vector gather: first board position whose id-channel == piece id
// ---------------------------------------------------------------------------
__global__ __launch_bounds__(256) void piece_gather(
    const float* __restrict__ board, float* __restrict__ out)
{
    size_t g = (size_t)blockIdx.x * 256 + threadIdx.x;   // img*32 + p
    if (g >= (size_t)NBATCH * 32) return;
    size_t img = g >> 5;
    int target = (int)(g & 31) + 1;
    const float* ids = board + (img * 10 + 9) * 64;
    int idx = 0; bool found = false;
    for (int pos = 0; pos < 64; ++pos) {
        if (!found && (int)ids[pos] == target) { idx = pos; found = true; }
    }
    float* o = out + g * 10;
    #pragma unroll
    for (int c = 0; c < 10; ++c)
        o[c] = found ? board[(img * 10 + c) * 64 + idx] : 0.0f;
}

// ---------------------------------------------------------------------------
extern "C" void kernel_launch(void* const* d_in, const int* in_sizes, int n_in,
                              void* d_out, int out_size, void* d_ws, size_t ws_size,
                              hipStream_t stream) {
    const float* board       = (const float*)d_in[0];
    const float* conv0_w     = (const float*)d_in[1];
    const float* conv0_scale = (const float*)d_in[2];
    const float* conv0_bias  = (const float*)d_in[3];
    const float* convs_w     = (const float*)d_in[4];
    const float* convs_scale = (const float*)d_in[5];
    const float* convs_bias  = (const float*)d_in[6];
    const float* polc_w      = (const float*)d_in[7];
    const float* polc_scale  = (const float*)d_in[8];
    const float* polc_bias   = (const float*)d_in[9];
    const float* valc_w      = (const float*)d_in[10];
    const float* valc_scale  = (const float*)d_in[11];
    const float* valc_bias   = (const float*)d_in[12];
    const float* pfc_w       = (const float*)d_in[13];
    const float* pfc_b       = (const float*)d_in[14];
    const float* vfc1_w      = (const float*)d_in[15];
    const float* vfc1_b      = (const float*)d_in[16];
    const float* vfc2_w      = (const float*)d_in[17];
    const float* vfc2_b      = (const float*)d_in[18];

    char* ws = (char*)d_ws;
    _Float16* yA    = (_Float16*)(ws);                                   // 64 MB
    _Float16* yB    = (_Float16*)(ws + (size_t)67108864);                // 64 MB
    float*    p_raw = (float*)(ws + (size_t)134217728);                  // 8 MB
    float*    v_raw = (float*)(ws + (size_t)142606336);                  // 4 MB
    float*    stats = (float*)(ws + (size_t)146800640);                  // 9*64 f32
    float*    gabe  = (float*)(ws + (size_t)146803200);                  // 10*64 f32
    _Float16* wpack = (_Float16*)(ws + (size_t)146805760);               // 313 KB

    float* out_pol   = (float*)d_out;
    float* out_val   = out_pol + (size_t)NBATCH * 750;
    float* out_piece = out_val + NBATCH;

    const float CNT = (float)NBATCH * 64.0f;

    init_ws<<<1, 256, 0, stream>>>(stats, gabe);
    pack_weights<<<(WPK_TOTAL + 255) / 256, 256, 0, stream>>>(conv0_w, convs_w, pfc_w, wpack);

    // conv0: 9 in-ch (of 10-ch board), identity gabe, no relu on input
    conv_wmma<0><<<NBATCH / 2, 256, 0, stream>>>(board, (const _Float16*)nullptr,
                                                 gabe, wpack + WPK_CONV0, yA, stats);
    bn_finalize<<<1, 32, 0, stream>>>(stats, conv0_scale, conv0_bias, 32, CNT, gabe + 64);

    _Float16* cur = yA;
    _Float16* nxt = yB;
    for (int i = 0; i < 6; ++i) {
        conv_wmma<1><<<NBATCH / 2, 256, 0, stream>>>((const float*)nullptr, cur,
                                                     gabe + 64 * (i + 1),
                                                     wpack + WPK_CONVS + (size_t)i * 9216,
                                                     nxt, stats + 64 * (i + 1));
        bn_finalize<<<1, 32, 0, stream>>>(stats + 64 * (i + 1),
                                          convs_scale + i * 32, convs_bias + i * 32,
                                          32, CNT, gabe + 64 * (i + 2));
        _Float16* t = cur; cur = nxt; nxt = t;
    }

    head_conv<<<(NBATCH * 64) / 256, 256, 0, stream>>>(cur, gabe + 64 * 7,
                                                       polc_w, valc_w, p_raw, v_raw,
                                                       stats + 64 * 7, stats + 64 * 8);
    bn_finalize<<<1, 32, 0, stream>>>(stats + 64 * 7, polc_scale, polc_bias, 2, CNT, gabe + 64 * 8);
    bn_finalize<<<1, 32, 0, stream>>>(stats + 64 * 8, valc_scale, valc_bias, 1, CNT, gabe + 64 * 9);

    policy_wmma<<<NBATCH / 16, 256, 0, stream>>>(p_raw, gabe + 64 * 8,
                                                 wpack + WPK_PFC, pfc_b, out_pol);
    value_fc<<<NBATCH, 128, 0, stream>>>(v_raw, gabe + 64 * 9, vfc1_w, vfc1_b, vfc2_w, vfc2_b, out_val);
    piece_gather<<<(NBATCH * 32) / 256, 256, 0, stream>>>(board, out_piece);
}
